// WTMambav2_80135499809160
// MI455X (gfx1250) — matrix operand order repaired and checked by reading, not compiled, not run on hardware
//
#include <hip/hip_runtime.h>

// ---------------------------------------------------------------------------
// WTMamba block for gfx1250 (MI455X, wave32, WMMA).
// Dense projections run on v_wmma_f32_16x16x32_f16 (f16 in, f32 accumulate).
// B operands are pre-packed into the WMMA B lane layout by their producers so
// the GEMM inner loop is pure b128 loads + wmma.
// ---------------------------------------------------------------------------

typedef __attribute__((ext_vector_type(16))) _Float16 v16h;
typedef __attribute__((ext_vector_type(8)))  _Float16 v8h;
typedef __attribute__((ext_vector_type(8)))  float    v8f;

#define Bn   4
#define Cn   64
#define Hn   128
#define Wn   128
#define Dm   192      // D_INNER
#define Ln   4096     // 64*64
#define P2n  4096
#define NST  16       // D_STATE
#define DTR  6        // DT_RANK

// fast activation helpers: single v_rcp_f32 instead of IEEE division chains
__device__ __forceinline__ float fast_rcp(float v) { return __builtin_amdgcn_rcpf(v); }
__device__ __forceinline__ float sigmf(float v) { return fast_rcp(1.f + __expf(-v)); }
__device__ __forceinline__ float siluf(float v) { return v * sigmf(v); }

// canonical flat spatial index for scan direction k at scan step l (h2=w2=64).
// self-inverse for every k (transpose, reversal, and their composition).
__device__ __forceinline__ int scan_index(int k, int l) {
  switch (k & 3) {
    case 0:  return l;
    case 1:  return ((l & 63) << 6) | (l >> 6);
    case 2:  return 4095 - l;
    default: { int m = 4095 - l; return ((m & 63) << 6) | (m >> 6); }
  }
}

// Packed B-operand layout: per 32x16 (KxN) tile, 512 halfs; lane = khalf*16+ncol
// holds its 16 K-contiguous elements at [lane*16 .. lane*16+15].
// Tiles ordered [ntile][ktile] so one wave's K-sweep is contiguous.
__device__ __forceinline__ int packB(int k, int n, int ktiles) {
  int nt = n >> 4, kt = k >> 5;
  int lane = (((k >> 4) & 1) << 4) | (n & 15);
  return ((nt * ktiles + kt) * 32 + lane) * 16 + (k & 15);
}

__device__ __forceinline__ float wave_sum(float v) {
#pragma unroll
  for (int off = 16; off > 0; off >>= 1) v += __shfl_xor(v, off, 32);
  return v;
}

// ------------------------------ small utilities ----------------------------

__global__ void f2h_kernel(const float* __restrict__ in, _Float16* __restrict__ out, int n) {
  int i = blockIdx.x * 256 + threadIdx.x;
  if (i < n) out[i] = (_Float16)in[i];
}

// pad x_proj_w (4,38,192) -> (4,48,192) f16, zero rows 38..47
__global__ void padxproj_kernel(const float* __restrict__ in, _Float16* __restrict__ out) {
  int i = blockIdx.x * 256 + threadIdx.x;          // 4*48*192 = 36864
  if (i >= 4 * 48 * 192) return;
  int c = i % 192, rr = (i / 192) % 48, k = i / (48 * 192);
  out[i] = (rr < 38) ? (_Float16)in[(k * 38 + rr) * 192 + c] : (_Float16)0.f;
}

// ------------------------------ forward Haar DWT ---------------------------
// writes lowH (packed, K=64) and highH (packed, K=192), scales applied.
__global__ void dwt_kernel(const float* __restrict__ x, const float* __restrict__ wls,
                           const float* __restrict__ whs,
                           _Float16* __restrict__ lowH, _Float16* __restrict__ highH) {
  int idx = blockIdx.x * 256 + threadIdx.x;        // B*C*4096
  int p = idx & 4095, bc = idx >> 12, c = bc & 63, b = bc >> 6;
  int i = p >> 6, j = p & 63;
  const float* xb = x + ((long long)bc << 14);
  int r0 = (i * 2) * 128 + j * 2;
  float x00 = xb[r0], x01 = xb[r0 + 1], x10 = xb[r0 + 128], x11 = xb[r0 + 129];
  float y0 = 0.5f * (x00 + x01 + x10 + x11);
  float y1 = 0.5f * (x00 + x01 - x10 - x11);
  float y2 = 0.5f * (x00 - x01 + x10 - x11);
  float y3 = 0.5f * (x00 - x01 - x10 + x11);
  lowH[(long long)b * (Cn * P2n) + packB(c, p, 2)] = (_Float16)(y0 * wls[c]);
  int ch = c * 3;
  long long hb = (long long)b * (Dm * Ln);
  highH[hb + packB(ch + 0, p, 6)] = (_Float16)(y1 * whs[ch + 0]);
  highH[hb + packB(ch + 1, p, 6)] = (_Float16)(y2 * whs[ch + 1]);
  highH[hb + packB(ch + 2, p, 6)] = (_Float16)(y3 * whs[ch + 2]);
}

// ------------------------------ WMMA GEMM ----------------------------------
// D[m,n] = sum_k A[m,k] * B[k,n]. One wave per 16x16 tile, K fully unrolled.
// A: row-major f16 (reduction dim contiguous -> two b128 loads per step).
// B: packed tile layout (two b128 loads per step). AKD: A indexed by z&3.
template <int ACT, int KDIM, int AKD>
__global__ void gemm_wmma(const _Float16* __restrict__ A, const _Float16* __restrict__ Bp,
                          float* __restrict__ Dmat, int lda, int ldd,
                          long long Astride, long long Bstride, long long Dstride) {
  constexpr int KT = KDIM / 32;
  const int lane = threadIdx.x;       // 0..31
  const int half = lane >> 4;
  const int l15  = lane & 15;
  const int nt = blockIdx.x * 4 + threadIdx.y;
  const int n0 = nt * 16;
  const int m0 = blockIdx.y * 16;
  const int z  = blockIdx.z;

  const _Float16* Ab = A + (long long)(AKD ? (z & 3) : z) * Astride;
  const _Float16* Bb = Bp + (long long)z * Bstride + ((long long)(nt * KT) << 9) + (lane << 4);
  float* Db = Dmat + (long long)z * Dstride;

  __builtin_prefetch(Bb, 0, 1);       // global_prefetch_b8: pull K-sweep into cache

  v8f acc = {0.f, 0.f, 0.f, 0.f, 0.f, 0.f, 0.f, 0.f};
#pragma unroll
  for (int kt = 0; kt < KT; ++kt) {
    const int k0 = kt * 32;
    // A 16x32 operand: lane row m0+l15; halfs [0..15] at k0+half*16,
    // (two contiguous b128 chunks).
    const _Float16* Arow = Ab + (long long)(m0 + l15) * lda + k0 + half * 16;
    v8h a0 = *(const v8h*)(Arow);
    v8h a1 = *(const v8h*)(Arow + 8);
    v16h a = __builtin_shufflevector(a0, a1, 0, 1, 2, 3, 4, 5, 6, 7,
                                     8, 9, 10, 11, 12, 13, 14, 15);
    // B packed: lane's 16 halfs contiguous.
    const v8h* Bl = (const v8h*)(Bb + (kt << 9));
    v8h b0 = Bl[0];
    v8h b1 = Bl[1];
    v16h bv = __builtin_shufflevector(b0, b1, 0, 1, 2, 3, 4, 5, 6, 7,
                                      8, 9, 10, 11, 12, 13, 14, 15);
    acc = __builtin_amdgcn_wmma_f32_16x16x32_f16(false, a, false, bv, (short)0, acc,
                                                 false, false);
  }
  // C/D layout: VGPR v -> row m0 + v + 8*half, col n0 + l15
#pragma unroll
  for (int v = 0; v < 8; ++v) {
    float val = acc[v];
    if (ACT == 1) val = siluf(val);
    Db[(long long)(m0 + v + half * 8) * ldd + (n0 + l15)] = val;
  }
}

// ------------------------------ depthwise conv 3x3 + SiLU ------------------
// writes x1c (linear f32, consumed by scan/LN) and xsPacked: 4 scan-direction
// packed f16 copies (column permutation folded in via the involution).
__global__ void dwconv_kernel(const float* __restrict__ xz, const float* __restrict__ w,
                              const float* __restrict__ bias, float* __restrict__ x1c,
                              _Float16* __restrict__ xsPacked) {
  int idx = blockIdx.x * 256 + threadIdx.x;        // B*192*4096
  int p = idx & 4095, bd = idx >> 12, d = bd % Dm, b = bd / Dm;
  int i = p >> 6, j = p & 63;
  const float* src = xz + ((long long)(b * 2 * Dm + d) << 12);
  float acc = bias[d];
#pragma unroll
  for (int ky = 0; ky < 3; ++ky) {
    int ii = i + ky - 1;
    if ((unsigned)ii >= 64u) continue;
#pragma unroll
    for (int kx = 0; kx < 3; ++kx) {
      int jj = j + kx - 1;
      if ((unsigned)jj >= 64u) continue;
      acc += w[d * 9 + ky * 3 + kx] * src[ii * 64 + jj];
    }
  }
  float o = siluf(acc);
  x1c[idx] = o;
  _Float16 oh = (_Float16)o;
#pragma unroll
  for (int kd = 0; kd < 4; ++kd) {
    int n = scan_index(kd, p);                     // column where p appears
    xsPacked[(long long)(b * 4 + kd) * (Dm * Ln) + packB(d, n, 6)] = oh;
  }
}

// ------------------------------ selective scan -----------------------------
// one block per (b,k): 192 threads (one per d), 16-state recurrence in regs,
// 4096 sequential steps; contributions accumulated at canonical positions.
__global__ void scan_kernel(const float* __restrict__ xdbl, const float* __restrict__ x1c,
                            const float* __restrict__ dtw_g, const float* __restrict__ dtb_g,
                            const float* __restrict__ A_log, float* __restrict__ ymerge) {
  const int z = blockIdx.x, b = z >> 2, k = z & 3;
  const int d = threadIdx.x;                       // 0..191
  __shared__ float sh[38];                          // dts[6] | Bs[16] | Cs[16]
  float dtw[DTR];
#pragma unroll
  for (int r = 0; r < DTR; ++r) dtw[r] = dtw_g[(k * Dm + d) * DTR + r];
  const float dtb = dtb_g[k * Dm + d];
  float Ar[NST], hst[NST];
#pragma unroll
  for (int n = 0; n < NST; ++n) {
    Ar[n] = -__expf(A_log[(k * Dm + d) * NST + n]);
    hst[n] = 0.f;
  }
  const float* base = xdbl + (long long)z * 48 * Ln;
  const float* urow = x1c + (long long)(b * Dm + d) * Ln;
  float* yrow = ymerge + (long long)(b * Dm + d) * Ln;
  for (int l = 0; l < Ln; ++l) {
    if (d < 38) sh[d] = base[(long long)d * Ln + l];
    __syncthreads();
    float dt = dtb;
#pragma unroll
    for (int r = 0; r < DTR; ++r) dt += dtw[r] * sh[r];
    float delta = (dt > 20.f) ? dt : log1pf(__expf(dt));   // softplus
    int pos = scan_index(k, l);
    float du = delta * urow[pos];
    float y = 0.f;
#pragma unroll
    for (int n = 0; n < NST; ++n) {
      hst[n] = hst[n] * __expf(delta * Ar[n]) + du * sh[DTR + n];
      y += hst[n] * sh[DTR + NST + n];
    }
    atomicAdd(&yrow[pos], y);
    __syncthreads();
  }
}

// ------------------------------ LN over d + SiLU(z) gate -------------------
// one wave per position; merged y += x1c * sum_k Ds[k,d] folded in here.
// output written packed (B operand of out_proj GEMM).
__global__ void ln_gate_kernel(const float* __restrict__ ymerge, const float* __restrict__ x1c,
                               const float* __restrict__ xz, const float* __restrict__ Ds,
                               const float* __restrict__ ons, const float* __restrict__ onb,
                               _Float16* __restrict__ gatedH) {
  int lane = threadIdx.x & 31, wv = threadIdx.x >> 5;
  int pg = blockIdx.x * 8 + wv;                    // B*L positions
  int b = pg >> 12, p = pg & 4095;
  float vals[6], s = 0.f, sq = 0.f;
#pragma unroll
  for (int i = 0; i < 6; ++i) {
    int d = lane + (i << 5);
    float sds = Ds[d] + Ds[Dm + d] + Ds[2 * Dm + d] + Ds[3 * Dm + d];
    long long o = ((long long)(b * Dm + d) << 12) + p;
    float v = ymerge[o] + x1c[o] * sds;
    vals[i] = v; s += v; sq += v * v;
  }
  s = wave_sum(s); sq = wave_sum(sq);
  float mean = s * (1.f / 192.f);
  float var  = sq * (1.f / 192.f) - mean * mean;
  float inv  = rsqrtf(var + 1e-5f);
#pragma unroll
  for (int i = 0; i < 6; ++i) {
    int d = lane + (i << 5);
    float zn = xz[((long long)(b * 2 * Dm + Dm + d) << 12) + p];
    float g = (vals[i] - mean) * inv * ons[d] + onb[d];
    g *= siluf(zn);
    gatedH[(long long)b * (Dm * Ln) + packB(d, p, 6)] = (_Float16)g;
  }
}

// ------------------------------ LN2 over 3C=192 channels, in place ---------
__global__ void ln2_kernel(float* __restrict__ hbuf, const float* __restrict__ s2,
                           const float* __restrict__ b2) {
  int lane = threadIdx.x & 31, wv = threadIdx.x >> 5;
  int pg = blockIdx.x * 8 + wv;
  int b = pg >> 12, p = pg & 4095;
  float vals[6], s = 0.f, sq = 0.f;
#pragma unroll
  for (int i = 0; i < 6; ++i) {
    int d = lane + (i << 5);
    float v = hbuf[((long long)(b * Dm + d) << 12) + p];
    vals[i] = v; s += v; sq += v * v;
  }
  s = wave_sum(s); sq = wave_sum(sq);
  float mean = s * (1.f / 192.f);
  float inv  = rsqrtf(sq * (1.f / 192.f) - mean * mean + 1e-5f);
#pragma unroll
  for (int i = 0; i < 6; ++i) {
    int d = lane + (i << 5);
    hbuf[((long long)(b * Dm + d) << 12) + p] = (vals[i] - mean) * inv * s2[d] + b2[d];
  }
}

// ------------------------------ merge + inverse Haar DWT -------------------
__global__ void iwt_kernel(const float* __restrict__ lowAct, const float* __restrict__ high,
                           float* __restrict__ yfull) {
  int idx = blockIdx.x * 256 + threadIdx.x;        // B*C*4096
  int p = idx & 4095, bc = idx >> 12, c = bc & 63, b = bc >> 6;
  int i = p >> 6, j = p & 63;
  float m[4];
#pragma unroll
  for (int s = 0; s < 4; ++s) {
    int q = c * 4 + s;                             // reference channel scramble
    if (q < 64) {
      m[s] = lowAct[((long long)(b * 64 + q) << 12) + p];
    } else {
      int hc = q - 64;
      m[s] = lowAct[((long long)(b * 64 + (hc & 63)) << 12) + p] *
             high[((long long)(b * Dm + hc) << 12) + p];
    }
  }
  float* ob = yfull + ((long long)bc << 14);
  int r = (i * 2) * 128 + j * 2;
  ob[r]       = 0.5f * (m[0] + m[1] + m[2] + m[3]);
  ob[r + 1]   = 0.5f * (m[0] + m[1] - m[2] - m[3]);
  ob[r + 128] = 0.5f * (m[0] - m[1] + m[2] - m[3]);
  ob[r + 129] = 0.5f * (m[0] - m[1] - m[2] + m[3]);
}

// ------------------------------ ECA + residual -----------------------------
__global__ void pool_kernel(const float* __restrict__ yfull, float* __restrict__ pooled) {
  __shared__ float sm[256];
  int bc = blockIdx.x;
  const float* src = yfull + ((long long)bc << 14);
  float s = 0.f;
  for (int i = threadIdx.x; i < 16384; i += 256) s += src[i];
  sm[threadIdx.x] = s;
  __syncthreads();
  for (int off = 128; off > 0; off >>= 1) {
    if (threadIdx.x < off) sm[threadIdx.x] += sm[threadIdx.x + off];
    __syncthreads();
  }
  if (threadIdx.x == 0) pooled[bc] = sm[0] * (1.f / 16384.f);
}

__global__ void eca_kernel(const float* __restrict__ pooled, const float* __restrict__ w,
                           float* __restrict__ att) {
  int t = threadIdx.x;                             // 256 = B*C
  if (t >= Bn * Cn) return;
  int b = t >> 6, c = t & 63;
  float acc = 0.f;
#pragma unroll
  for (int i = 0; i < 5; ++i) {
    int j = c + i - 2;
    if ((unsigned)j < 64u) acc += w[i] * pooled[b * 64 + j];
  }
  att[t] = sigmf(acc);
}

__global__ void final_kernel(const float* __restrict__ yfull, const float* __restrict__ att,
                             const float* __restrict__ x, float* __restrict__ out) {
  int idx = blockIdx.x * 256 + threadIdx.x;        // B*C*H*W
  int bc = idx >> 14;
  out[idx] = yfull[idx] * att[bc] + x[idx];
}

// ---------------------------------------------------------------------------
extern "C" void kernel_launch(void* const* d_in, const int* in_sizes, int n_in,
                              void* d_out, int out_size, void* d_ws, size_t ws_size,
                              hipStream_t stream) {
  (void)in_sizes; (void)n_in; (void)out_size; (void)ws_size;
  const float* x         = (const float*)d_in[0];
  const float* wls       = (const float*)d_in[1];
  const float* whs       = (const float*)d_in[2];
  const float* low_w     = (const float*)d_in[3];   // (C,C,1,1)
  const float* inproj_w  = (const float*)d_in[4];   // (384,192)
  const float* dw_w      = (const float*)d_in[5];   // (192,1,3,3)
  const float* dw_b      = (const float*)d_in[6];
  const float* xproj_w   = (const float*)d_in[7];   // (4,38,192)
  const float* dtp_w     = (const float*)d_in[8];   // (4,192,6)
  const float* dtp_b     = (const float*)d_in[9];
  const float* A_log     = (const float*)d_in[10];
  const float* Ds        = (const float*)d_in[11];
  const float* ons       = (const float*)d_in[12];
  const float* onb       = (const float*)d_in[13];
  const float* outproj_w = (const float*)d_in[14];  // (192,192)
  const float* ln2s      = (const float*)d_in[15];
  const float* ln2b      = (const float*)d_in[16];
  const float* eca       = (const float*)d_in[17];
  float* outp = (float*)d_out;

  char* ws = (char*)d_ws;
  size_t off = 0;
  auto take = [&](size_t bytes) -> char* {
    char* p = ws + off;
    off += (bytes + 255) & ~(size_t)255;
    return p;
  };

  const size_t NLOW  = (size_t)Bn * Cn * P2n;       // 1,048,576
  const size_t NHIGH = (size_t)Bn * Dm * Ln;        // 3,145,728
  const size_t NXZ   = (size_t)Bn * 2 * Dm * Ln;    // 6,291,456
  const size_t NXDBL = (size_t)16 * 48 * Ln;        // 3,145,728
  const size_t NFULL = (size_t)Bn * Cn * Hn * Wn;   // 4,194,304

  _Float16* lowH     = (_Float16*)take(NLOW * 2);        // packed K=64
  _Float16* highH    = (_Float16*)take(NHIGH * 2);       // packed K=192
  _Float16* inprojH  = (_Float16*)take(384 * 192 * 2);
  _Float16* lowcH    = (_Float16*)take(64 * 64 * 2);
  _Float16* xprojH   = (_Float16*)take(4 * 48 * 192 * 2);
  _Float16* outprojH = (_Float16*)take(192 * 192 * 2);
  float*    lowAct   = (float*)take(NLOW * 4);
  float*    xz       = (float*)take(NXZ * 4);
  float*    x1c      = (float*)take(NHIGH * 4);
  _Float16* xsPacked = (_Float16*)take(NHIGH * 4 * 2);   // 4 directions, packed
  float*    xdbl     = (float*)take(NXDBL * 4);
  float*    ymerge   = (float*)take(NHIGH * 4);
  _Float16* gatedH   = (_Float16*)take(NHIGH * 2);       // packed K=192
  float*    highOut  = (float*)take(NHIGH * 4);
  float*    yfull    = (float*)take(NFULL * 4);
  float*    pooled   = (float*)take(Bn * Cn * 4);
  float*    att      = (float*)take(Bn * Cn * 4);

  // 1) weight conversions to f16
  f2h_kernel<<<(384 * 192 + 255) / 256, 256, 0, stream>>>(inproj_w, inprojH, 384 * 192);
  f2h_kernel<<<(64 * 64 + 255) / 256, 256, 0, stream>>>(low_w, lowcH, 64 * 64);
  f2h_kernel<<<(192 * 192 + 255) / 256, 256, 0, stream>>>(outproj_w, outprojH, 192 * 192);
  padxproj_kernel<<<(4 * 48 * 192 + 255) / 256, 256, 0, stream>>>(xproj_w, xprojH);

  // 2) forward Haar DWT + per-channel scales (packed B operands)
  dwt_kernel<<<NLOW / 256, 256, 0, stream>>>(x, wls, whs, lowH, highH);

  // 3) low path: 1x1 conv GEMM (M=64,K=64,N=4096 per batch) + SiLU
  gemm_wmma<1, 64, 0><<<dim3(64, 4, Bn), dim3(32, 4), 0, stream>>>(
      lowcH, lowH, lowAct, 64, P2n, 0, (long long)64 * P2n, (long long)64 * P2n);

  // 4) in_proj GEMM (M=384,K=192,N=4096 per batch)
  gemm_wmma<0, 192, 0><<<dim3(64, 24, Bn), dim3(32, 4), 0, stream>>>(
      inprojH, highH, xz, 192, Ln, 0, (long long)Dm * Ln, (long long)2 * Dm * Ln);

  // 5) depthwise 3x3 + bias + SiLU; scatter into 4 direction-packed copies
  dwconv_kernel<<<NHIGH / 256, 256, 0, stream>>>(xz, dw_w, dw_b, x1c, xsPacked);

  // 6) x_proj GEMM (M=48 padded, K=192, N=4096, 16 batches z=b*4+kdir)
  gemm_wmma<0, 192, 1><<<dim3(64, 3, 16), dim3(32, 4), 0, stream>>>(
      xprojH, xsPacked, xdbl, 192, Ln, (long long)48 * 192, (long long)Dm * Ln,
      (long long)48 * Ln);

  // 7) selective scan (delta folded in), atomically merged over 4 directions
  hipMemsetAsync(ymerge, 0, NHIGH * 4, stream);
  scan_kernel<<<Bn * 4, Dm, 0, stream>>>(xdbl, x1c, dtp_w, dtp_b, A_log, ymerge);

  // 8) LN over d + SiLU(z) gating -> packed f16 for out_proj
  ln_gate_kernel<<<(Bn * Ln) / 8, 256, 0, stream>>>(ymerge, x1c, xz, Ds, ons, onb, gatedH);

  // 9) out_proj GEMM (M=192,K=192,N=4096 per batch)
  gemm_wmma<0, 192, 0><<<dim3(64, 12, Bn), dim3(32, 4), 0, stream>>>(
      outprojH, gatedH, highOut, 192, Ln, 0, (long long)Dm * Ln, (long long)Dm * Ln);

  // 10) LN2 over 192 channels (in place)
  ln2_kernel<<<(Bn * Ln) / 8, 256, 0, stream>>>(highOut, ln2s, ln2b);

  // 11) merge (low, low*high) per reference channel scramble + inverse DWT
  iwt_kernel<<<NLOW / 256, 256, 0, stream>>>(lowAct, highOut, yfull);

  // 12) ECA attention + residual
  pool_kernel<<<Bn * Cn, 256, 0, stream>>>(yfull, pooled);
  eca_kernel<<<1, 256, 0, stream>>>(pooled, eca, att);
  final_kernel<<<NFULL / 256, 256, 0, stream>>>(yfull, att, x, outp);
}
